// LmLSTMSharedEmb_70016556860045
// MI455X (gfx1250) — matrix operand order repaired and checked by reading, not compiled, not run on hardware
//
#include <hip/hip_runtime.h>
#include <hip/hip_bf16.h>

#define V_SZ 32000
#define H_SZ 512
#define DEPTH_N 2
#define B_SZ 8
#define S_SZ 512
#define G4 (4 * H_SZ)        // 2048
#define M_SZ (B_SZ * S_SZ)   // 4096

typedef __attribute__((ext_vector_type(16))) __bf16 v16bf;
typedef __attribute__((ext_vector_type(8)))  float  v8f;
typedef __attribute__((ext_vector_type(4)))  unsigned int u32x4;
typedef __attribute__((ext_vector_type(8)))  int i32x8;
typedef __attribute__((ext_vector_type(4)))  int i32x4;

__device__ __forceinline__ float sigf(float x) { return 1.0f / (1.0f + __expf(-x)); }

// A-fragment (16x32 bf16, contiguous k = 16*hi + e): lane holds row (lane&15); rowbase points at A[row] + k0.
__device__ __forceinline__ v16bf load_a16x32(const __bf16* rowbase, int hi) {
  v16bf a;
#pragma unroll
  for (int e = 0; e < 8; ++e) a[e] = rowbase[hi * 16 + e];
#pragma unroll
  for (int e = 0; e < 8; ++e) a[8 + e] = rowbase[hi * 16 + 8 + e];
  return a;
}

// B-fragment (32x16 bf16): lane holds col (lane&15); p = B_rowmajor(N x K) + n*ldb + k0 + hi*16.
__device__ __forceinline__ v16bf load_b32x16(const __bf16* p) {
  v16bf b;
#pragma unroll
  for (int e = 0; e < 16; ++e) b[e] = p[e];
  return b;
}

// ---- TDM stage: copy contiguous 16x512 bf16 A tile (16 KB) from global into LDS offset 0 ----
// 1-D tile descriptor: data_size=2B, tensor_dim0 = tile_dim0 = 8192 elems, tile_dim1 = 0.
__device__ __forceinline__ void stage_a_tile(const __bf16* gsrc, __bf16* lds_dst,
                                             int wave, int tid, int nthreads) {
#if __has_builtin(__builtin_amdgcn_tensor_load_to_lds) && __has_builtin(__builtin_amdgcn_s_wait_tensorcnt)
  (void)lds_dst; (void)tid; (void)nthreads;
  if (wave == 0) {
    unsigned long long ga = (unsigned long long)gsrc;
    const int ELEMS = 16 * H_SZ;                    // 8192 bf16 elements
    u32x4 g0;
    g0[0] = 1u;                                     // count=1, user-mode, no gather
    g0[1] = 0u;                                     // lds_addr = 0 (dynamic LDS base)
    g0[2] = (unsigned)(ga & 0xFFFFFFFFu);           // global_addr[31:0]
    g0[3] = (unsigned)((ga >> 32) & 0x01FFFFFFu) | (2u << 30);  // global_addr[56:32] | type=2
    i32x8 g1;
    g1[0] = (1 << 16);                              // data_size=1 (2 bytes), workgroup_mask=0
    g1[1] = (ELEMS & 0xFFFF) << 16;                 // tensor_dim0[15:0] @ bits 63:48
    g1[2] = ((ELEMS >> 16) & 0xFFFF) | (1 << 16);   // tensor_dim0[31:16] | tensor_dim1=1
    g1[3] = (ELEMS & 0xFFFF) << 16;                 // tensor_dim1 hi=0 | tile_dim0=8192
    g1[4] = 0;                                      // tile_dim1=0 (1-D), tile_dim2=0
    g1[5] = ELEMS;                                  // tensor_dim0_stride[31:0]
    g1[6] = 0;
    g1[7] = 0;
    i32x4 gz4;
    gz4[0] = 0; gz4[1] = 0; gz4[2] = 0; gz4[3] = 0;
    i32x8 gz8;
    gz8[0] = 0; gz8[1] = 0; gz8[2] = 0; gz8[3] = 0;
    gz8[4] = 0; gz8[5] = 0; gz8[6] = 0; gz8[7] = 0;
    __builtin_amdgcn_tensor_load_to_lds(g0, g1, gz4, gz4, gz8, 0);  // 6-arg (clang-23) form
    __builtin_amdgcn_s_wait_tensorcnt(0);
  }
  __syncthreads();
#else
  (void)wave;
  for (int i = tid; i < 16 * H_SZ; i += nthreads) lds_dst[i] = gsrc[i];
  __syncthreads();
#endif
}

// ---------------- embedding gather + mask ----------------
__global__ void embed_kernel(const long long* __restrict__ tok,
                             const float* __restrict__ emb,
                             __bf16* __restrict__ x, float* __restrict__ fmask) {
  int bs = blockIdx.x;                       // bs = b*S + s
  long long t = tok[bs];
  if (threadIdx.x == 0) fmask[bs] = (t != 0) ? 1.0f : 0.0f;
  const float* src = emb + (size_t)t * H_SZ;
  __bf16* dst = x + (size_t)bs * H_SZ;
  for (int h = threadIdx.x; h < H_SZ; h += blockDim.x) dst[h] = (__bf16)src[h];
}

// ---------------- transpose W (DEPTH,H,4H) f32 -> (DEPTH,4H,H) bf16 ----------------
__global__ void wtrans_kernel(const float* __restrict__ W, __bf16* __restrict__ WT, int total) {
  int idx = blockIdx.x * blockDim.x + threadIdx.x;
  if (idx >= total) return;
  int k = idx % H_SZ;
  int n = (idx / H_SZ) % G4;
  int l = idx / (H_SZ * G4);
  WT[idx] = (__bf16)W[((size_t)l * H_SZ + k) * G4 + n];
}

// ---------------- f32 -> bf16 convert ----------------
__global__ void cvt_bf16_kernel(const float* __restrict__ src, __bf16* __restrict__ dst, size_t n) {
  size_t i = blockIdx.x * (size_t)blockDim.x + threadIdx.x;
  if (i < n) dst[i] = (__bf16)src[i];
}

// ---------------- xproj GEMM: (4096x512 bf16) @ WxT(2048x512 bf16)^T + bias -> (s,b,2048) f32 --
__global__ void gemm_xproj_kernel(const __bf16* __restrict__ A, const __bf16* __restrict__ Bm,
                                  const float* __restrict__ bias, float* __restrict__ xproj) {
  extern __shared__ __bf16 atile[];            // 16 x 512
  int lane = threadIdx.x & 31, wave = threadIdx.x >> 5;
  int m0 = blockIdx.y * 16;
  int n0 = blockIdx.x * 256 + wave * 64;
  int mlane = lane & 15, hi = lane >> 4;
  stage_a_tile(A + (size_t)m0 * H_SZ, atile, wave, threadIdx.x, blockDim.x);
  const __bf16* arow = atile + (size_t)mlane * H_SZ;
  v8f acc[4] = {};
  for (int k0 = 0; k0 < H_SZ; k0 += 32) {
    v16bf a = load_a16x32(arow + k0, hi);
#pragma unroll
    for (int nt = 0; nt < 4; ++nt) {
      const __bf16* bp = Bm + (size_t)(n0 + nt * 16 + mlane) * H_SZ + k0 + hi * 16;
      __builtin_prefetch(bp + 32, 0, 3);       // global_prefetch_b8: next K slab
      v16bf bf = load_b32x16(bp);
      acc[nt] = __builtin_amdgcn_wmma_f32_16x16x32_bf16(false, a, false, bf, (short)0,
                                                        acc[nt], false, false);
    }
  }
#pragma unroll
  for (int nt = 0; nt < 4; ++nt) {
    int col = n0 + nt * 16 + mlane;
    float bb = bias[col];
#pragma unroll
    for (int r = 0; r < 8; ++r) {
      int m = m0 + r + 8 * hi;
      int bb_idx = m >> 9, s = m & (S_SZ - 1);
      xproj[((size_t)(s * B_SZ + bb_idx)) * G4 + col] = acc[nt][r] + bb;
    }
  }
}

// ---------------- persistent LSTM scan: 1 WG, 32 waves, h in LDS, WMMA recurrent GEMM -------
__global__ void __launch_bounds__(1024) lstm_scan_kernel(const float* __restrict__ xproj,
                                                         const __bf16* __restrict__ WhT,
                                                         const float* __restrict__ fmask,
                                                         __bf16* __restrict__ xout) {
  __shared__ __bf16 h_lds[16][H_SZ];   // rows 8..15 stay zero (M=16 pad for wmma)
  int tid = threadIdx.x;
  int lane = tid & 31, wave = tid >> 5;        // 32 waves
  int mlane = lane & 15, hi = lane >> 4;
  for (int i = tid; i < 16 * H_SZ; i += 1024) ((__bf16*)h_lds)[i] = (__bf16)0.0f;
  float cst[8];
#pragma unroll
  for (int r = 0; r < 8; ++r) cst[r] = 0.0f;
  int c0 = wave * 16;
  int col = c0 + mlane;                        // hidden column owned by this lane
  __syncthreads();

  for (int t = 0; t < S_SZ; ++t) {
    v8f z[4] = {};                             // i, f, g, o tiles (16x16 each, f32 accum)
    for (int k0 = 0; k0 < H_SZ; k0 += 32) {
      v16bf a = load_a16x32(&h_lds[mlane][k0], hi);
#pragma unroll
      for (int g = 0; g < 4; ++g) {
        const __bf16* bp = WhT + (size_t)(g * H_SZ + col) * H_SZ + k0 + hi * 16;
        v16bf bf = load_b32x16(bp);
        z[g] = __builtin_amdgcn_wmma_f32_16x16x32_bf16(false, a, false, bf, (short)0,
                                                       z[g], false, false);
      }
    }
    float hnew[8];
#pragma unroll
    for (int r = 0; r < 8; ++r) {
      int brow = r + 8 * hi;
      float x0 = 0.f, x1 = 0.f, x2 = 0.f, x3 = 0.f, fm = 0.f, hold = 0.f;
      if (brow < B_SZ) {
        size_t base = ((size_t)(t * B_SZ + brow)) * G4;
        x0 = xproj[base + 0 * H_SZ + col];
        x1 = xproj[base + 1 * H_SZ + col];
        x2 = xproj[base + 2 * H_SZ + col];
        x3 = xproj[base + 3 * H_SZ + col];
        fm = fmask[brow * S_SZ + t];
        hold = (float)h_lds[r][col];
      }
      float zi = z[0][r] + x0, zf = z[1][r] + x1, zg = z[2][r] + x2, zo = z[3][r] + x3;
      float cn = sigf(zf) * cst[r] + sigf(zi) * tanhf(zg);
      float hn = sigf(zo) * tanhf(cn);
      cst[r] = fm * cn + (1.0f - fm) * cst[r];
      hnew[r] = fm * hn + (1.0f - fm) * hold;
    }
    __syncthreads();                           // all reads of h done before overwrite
    if (hi == 0) {
#pragma unroll
      for (int r = 0; r < 8; ++r) {
        __bf16 hb = (__bf16)hnew[r];
        h_lds[r][col] = hb;
        xout[((size_t)r * S_SZ + t) * H_SZ + col] = hb;
      }
    }
    __syncthreads();                           // h visible for next step
  }
}

// ---------------- logits GEMM: x(4096x512 bf16) @ emb_bf16(32000x512)^T, fused mask ---------
__global__ void gemm_logits_kernel(const __bf16* __restrict__ A, const __bf16* __restrict__ Bm,
                                   const float* __restrict__ fmask, float* __restrict__ out) {
  extern __shared__ __bf16 atile[];            // 16 x 512
  int lane = threadIdx.x & 31, wave = threadIdx.x >> 5;
  int m0 = blockIdx.y * 16;
  int n0 = blockIdx.x * 256 + wave * 64;
  int mlane = lane & 15, hi = lane >> 4;
  stage_a_tile(A + (size_t)m0 * H_SZ, atile, wave, threadIdx.x, blockDim.x);
  const __bf16* arow = atile + (size_t)mlane * H_SZ;
  v8f acc[4] = {};
  for (int k0 = 0; k0 < H_SZ; k0 += 32) {
    v16bf a = load_a16x32(arow + k0, hi);
#pragma unroll
    for (int nt = 0; nt < 4; ++nt) {
      const __bf16* bp = Bm + (size_t)(n0 + nt * 16 + mlane) * H_SZ + k0 + hi * 16;
      __builtin_prefetch(bp + 32, 0, 3);       // global_prefetch_b8: next K slab
      v16bf bf = load_b32x16(bp);
      acc[nt] = __builtin_amdgcn_wmma_f32_16x16x32_bf16(false, a, false, bf, (short)0,
                                                        acc[nt], false, false);
    }
  }
  float fm[8];
#pragma unroll
  for (int r = 0; r < 8; ++r) fm[r] = fmask[m0 + r + 8 * hi];
#pragma unroll
  for (int nt = 0; nt < 4; ++nt) {
    int colv = n0 + nt * 16 + mlane;
    float add0 = (colv == 0) ? 1.0f : 0.0f;
#pragma unroll
    for (int r = 0; r < 8; ++r) {
      int m = m0 + r + 8 * hi;
      out[(size_t)m * V_SZ + colv] = acc[nt][r] * fm[r] + add0 * (1.0f - fm[r]);
    }
  }
}

extern "C" void kernel_launch(void* const* d_in, const int* in_sizes, int n_in,
                              void* d_out, int out_size, void* d_ws, size_t ws_size,
                              hipStream_t stream) {
  const long long* tokens = (const long long*)d_in[0];
  const float* emb  = (const float*)d_in[1];
  const float* Wx   = (const float*)d_in[2];
  const float* Wh   = (const float*)d_in[3];
  const float* bias = (const float*)d_in[4];
  float* out = (float*)d_out;

  char* ws = (char*)d_ws;
  size_t off = 0;
  auto take = [&](size_t bytes) { size_t o = off; off = (off + bytes + 255) & ~(size_t)255; return o; };
  float*  fmask  = (float*)(ws + take((size_t)M_SZ * 4));
  __bf16* xA     = (__bf16*)(ws + take((size_t)M_SZ * H_SZ * 2));
  __bf16* xB     = (__bf16*)(ws + take((size_t)M_SZ * H_SZ * 2));
  __bf16* WxT    = (__bf16*)(ws + take((size_t)DEPTH_N * G4 * H_SZ * 2));
  __bf16* WhT    = (__bf16*)(ws + take((size_t)DEPTH_N * G4 * H_SZ * 2));
  __bf16* embbf  = (__bf16*)(ws + take((size_t)V_SZ * H_SZ * 2));
  float*  xproj  = (float*)(ws + take((size_t)S_SZ * B_SZ * G4 * 4));
  (void)ws_size; (void)in_sizes; (void)n_in; (void)out_size;

  const size_t atile_bytes = (size_t)16 * H_SZ * sizeof(__bf16);  // 16 KB dynamic LDS

  // 1) gather embeddings -> bf16 x, compute mask
  embed_kernel<<<M_SZ, 256, 0, stream>>>(tokens, emb, xA, fmask);

  // 2) transpose + convert weights, convert shared embedding to bf16
  int wtot = DEPTH_N * G4 * H_SZ;
  wtrans_kernel<<<(wtot + 255) / 256, 256, 0, stream>>>(Wx, WxT, wtot);
  wtrans_kernel<<<(wtot + 255) / 256, 256, 0, stream>>>(Wh, WhT, wtot);
  size_t etot = (size_t)V_SZ * H_SZ;
  cvt_bf16_kernel<<<(unsigned)((etot + 255) / 256), 256, 0, stream>>>(emb, embbf, etot);

  // 3) two LSTM layers: batched xproj WMMA GEMM, then persistent WMMA scan
  __bf16* xin = xA;
  __bf16* xnext = xB;
  for (int layer = 0; layer < DEPTH_N; ++layer) {
    gemm_xproj_kernel<<<dim3(G4 / 256, M_SZ / 16), 128, atile_bytes, stream>>>(
        xin, WxT + (size_t)layer * G4 * H_SZ, bias + (size_t)layer * G4, xproj);
    lstm_scan_kernel<<<1, 1024, 0, stream>>>(
        xproj, WhT + (size_t)layer * G4 * H_SZ, fmask, xnext);
    __bf16* tmp = xin; xin = xnext; xnext = tmp;
  }

  // 4) logits = x @ emb^T with fused mask / one-hot fill
  gemm_logits_kernel<<<dim3(V_SZ / 256, M_SZ / 16), 128, atile_bytes, stream>>>(xin, embbf, fmask, out);
}
